// TACEncoder_21809843929187
// MI455X (gfx1250) — compile-verified, matrix operand order checked by reading, not run on hardware
//
#include <hip/hip_runtime.h>
#include <math.h>

typedef __attribute__((ext_vector_type(16))) _Float16 v16h;
typedef __attribute__((ext_vector_type(8)))  float    v8f;

#define DEV __device__ __forceinline__

// ---------------------------------------------------------------------------
// WMMA fragment loader (16x32 f16, row-major source with leading dim `ld`).
// Matches CDNA5 16-bit A-matrix layout; B uses the symmetric layout (pass the
// activation tile column-major so its "rows" are output columns).
// ---------------------------------------------------------------------------
DEV v16h load_frag(const _Float16* base, int ld, int lane) {
    int r = lane & 15;
    const _Float16* p = base + (size_t)r * ld + ((lane & 16) ? 8 : 0);
    v16h f;
#pragma unroll
    for (int i = 0; i < 8; ++i) f[i] = p[i];
#pragma unroll
    for (int i = 0; i < 8; ++i) f[8 + i] = p[16 + i];
    return f;
}

// One fused MLP layer over a 160-column LDS tile.
// src: LDS, column-major, column stride = KT*32 (f16, zero padded)
// W  : global f16, [MT*16][KT*32] row-major (zero padded)
// dst: LDS, column-major, column stride = dstld
// y = acc*scale[m] + shift[m]; optional LeakyReLU(0.2)
template<int KT, int MT>
DEV void mlp_layer(const _Float16* __restrict__ W, const float* __restrict__ scale,
                   const float* __restrict__ shift, const _Float16* src, _Float16* dst,
                   int dstld, bool lrelu, int lane, int wave)
{
    for (int job = wave; job < MT * 10; job += 8) {
        int mt = job / 10, ct = job % 10;
        v8f acc = {};
#pragma unroll
        for (int kt = 0; kt < KT; ++kt) {
            v16h a = load_frag(W + (size_t)(mt * 16) * (KT * 32) + kt * 32, KT * 32, lane);
            v16h b = load_frag(src + (size_t)(ct * 16) * (KT * 32) + kt * 32, KT * 32, lane);
            acc = __builtin_amdgcn_wmma_f32_16x16x32_f16(false, a, false, b, (short)0, acc,
                                                         false, false);
        }
        int col = ct * 16 + (lane & 15);
        int rb = mt * 16 + ((lane & 16) ? 8 : 0);
#pragma unroll
        for (int r = 0; r < 8; ++r) {
            int m = rb + r;
            float y = acc[r] * scale[m] + shift[m];
            if (lrelu) y = (y >= 0.f) ? y : 0.2f * y;
            dst[(size_t)col * dstld + m] = (_Float16)y;
        }
    }
}

// ---------------------------------------------------------------------------
// Fused EdgeConv: build edge features -> conv1(bn,lrelu) -> conv2(bn,lrelu)
// -> conv3(bias) -> max over K.  One block handles 8 points (160 columns).
// feat: [B, NPTS, CIN]  knn: [B, NPTS, 20]  out: [B, NPTS, COUT]
// ---------------------------------------------------------------------------
template<int CIN, int MID, int COUT, int NPTS>
__global__ __launch_bounds__(256) void edgeconv_kernel(
    const float* __restrict__ feat, const int* __restrict__ knn,
    const _Float16* __restrict__ W1, const float* __restrict__ s1, const float* __restrict__ h1,
    const _Float16* __restrict__ W2, const float* __restrict__ s2, const float* __restrict__ h2,
    const _Float16* __restrict__ W3, const float* __restrict__ s3, const float* __restrict__ h3,
    float* __restrict__ out)
{
    constexpr int TP = 8, KNB = 20, COLS = 160;
    constexpr int K1 = ((2 * CIN + 31) / 32) * 32;
    constexpr int K2 = MID;                     // MID is a multiple of 32
    constexpr int EROWS = (K1 > K2) ? K1 : K2;
    extern __shared__ char smem[];
    _Float16* bufE = (_Float16*)smem;           // EROWS x COLS (E, then H2)
    _Float16* bufH = bufE + EROWS * COLS;       // K2 x COLS    (H1)
    float*    bufO = (float*)(bufH + K2 * COLS); // 8 waves x 16 x COLS

    int b  = blockIdx.x / (NPTS / TP);
    int p0 = (blockIdx.x % (NPTS / TP)) * TP;
    const float* fb = feat + (size_t)b * NPTS * CIN;
    const int*   kb = knn + ((size_t)b * NPTS + p0) * KNB;
    int tid = threadIdx.x, lane = tid & 31, wave = tid >> 5;

    // build edge features: rows[0..CIN) = central-neigh, rows[CIN..2CIN) = central
    for (int col = tid; col < COLS; col += 256) {
        int p = col / KNB, j = col - p * KNB;
        int nb = kb[p * KNB + j];
        const float* fc = fb + (size_t)(p0 + p) * CIN;
        const float* fn = fb + (size_t)nb * CIN;
        _Float16* e = bufE + (size_t)col * K1;
        for (int c = 0; c < CIN; ++c) {
            float cv = fc[c];
            e[c] = (_Float16)(cv - fn[c]);
            e[CIN + c] = (_Float16)cv;
        }
        for (int c = 2 * CIN; c < K1; ++c) e[c] = (_Float16)0.f;
    }
    __syncthreads();
    mlp_layer<K1 / 32, MID / 16>(W1, s1, h1, bufE, bufH, K2, true, lane, wave);
    __syncthreads();
    mlp_layer<K2 / 32, MID / 16>(W2, s2, h2, bufH, bufE, K2, true, lane, wave);
    __syncthreads();

    // layer3 + max over K: each wave owns whole m-tiles -> no cross-wave sync
    float* myO = bufO + (size_t)wave * 16 * COLS;
    for (int mt = wave; mt < COUT / 16; mt += 8) {
        for (int ct = 0; ct < 10; ++ct) {
            v8f acc = {};
#pragma unroll
            for (int kt = 0; kt < K2 / 32; ++kt) {
                v16h a = load_frag(W3 + (size_t)(mt * 16) * K2 + kt * 32, K2, lane);
                v16h bb = load_frag(bufE + (size_t)(ct * 16) * K2 + kt * 32, K2, lane);
                acc = __builtin_amdgcn_wmma_f32_16x16x32_f16(false, a, false, bb, (short)0, acc,
                                                             false, false);
            }
            int col = ct * 16 + (lane & 15);
            int rb = (lane & 16) ? 8 : 0;
#pragma unroll
            for (int r = 0; r < 8; ++r) {
                int m = mt * 16 + rb + r;
                myO[(size_t)(rb + r) * COLS + col] = acc[r] * s3[m] + h3[m];
            }
        }
        __builtin_amdgcn_wave_barrier();  // LDS in-order within wave
        for (int t = lane; t < 16 * TP; t += 32) {
            int row = t & 15, p = t >> 4;
            float mx = -3.4e38f;
#pragma unroll
            for (int j = 0; j < KNB; ++j) mx = fmaxf(mx, myO[(size_t)row * COLS + p * KNB + j]);
            out[((size_t)b * NPTS + p0 + p) * COUT + mt * 16 + row] = mx;
        }
        __builtin_amdgcn_wave_barrier();
    }
}

// ---------------------------------------------------------------------------
// Fused AdaptGraphPooling MLPs + softmax + weighted sums.
// verts:[B,NV,3] feat:[B,NV,C] kp:[B,M,3] kf:[B,M,C] knn:[B,M,20]
// vout:[B,M,3]  fout:[B,M,C]
// ---------------------------------------------------------------------------
template<int C>
__global__ __launch_bounds__(256) void agp_kernel(
    const float* __restrict__ verts, const float* __restrict__ feat,
    const float* __restrict__ kp, const float* __restrict__ kf,
    const int* __restrict__ knn,
    const _Float16* __restrict__ PW1, const float* __restrict__ ps1, const float* __restrict__ ph1,
    const _Float16* __restrict__ PW2, const float* __restrict__ ps2, const float* __restrict__ ph2,
    const _Float16* __restrict__ AW1, const float* __restrict__ as1, const float* __restrict__ ah1,
    const _Float16* __restrict__ AW2, const float* __restrict__ as2, const float* __restrict__ ah2,
    float* __restrict__ vout, float* __restrict__ fout, int NV, int M)
{
    constexpr int TP = 8, KNB = 20, COLS = 160;
    constexpr int OC = 3 + C;
    constexpr int OCP = ((OC + 15) / 16) * 16;
    extern __shared__ char smem[];
    _Float16* P0  = (_Float16*)smem;            // 32  x COLS
    _Float16* H1  = P0 + 32 * COLS;             // 64  x COLS (pos mid / attn mid)
    _Float16* PE  = H1 + 64 * COLS;             // C   x COLS
    _Float16* A0  = PE + (size_t)C * COLS;      // C   x COLS
    _Float16* ATT = A0 + (size_t)C * COLS;      // OCP x COLS

    int b  = blockIdx.x / (M / TP);
    int p0 = (blockIdx.x % (M / TP)) * TP;
    const float* vb  = verts + (size_t)b * NV * 3;
    const float* fbb = feat + (size_t)b * NV * C;
    const float* kpb = kp + ((size_t)b * M + p0) * 3;
    const float* kfb = kf + ((size_t)b * M + p0) * C;
    const int*   kb  = knn + ((size_t)b * M + p0) * KNB;
    int tid = threadIdx.x, lane = tid & 31, wave = tid >> 5;

    // pos-MLP input: key_point - gathered vertices (3 rows, zero padded to 32)
    for (int col = tid; col < COLS; col += 256) {
        int p = col / KNB, j = col - p * KNB;
        int nb = kb[p * KNB + j];
        _Float16* e = P0 + (size_t)col * 32;
#pragma unroll
        for (int c = 0; c < 3; ++c) e[c] = (_Float16)(kpb[p * 3 + c] - vb[(size_t)nb * 3 + c]);
#pragma unroll
        for (int c = 3; c < 32; ++c) e[c] = (_Float16)0.f;
    }
    __syncthreads();
    mlp_layer<1, 4>(PW1, ps1, ph1, P0, H1, 64, true, lane, wave);        // 3->64 bn+lrelu
    __syncthreads();
    mlp_layer<2, C / 16>(PW2, ps2, ph2, H1, PE, C, false, lane, wave);   // 64->C bias
    __syncthreads();
    // attn input: (key_feat - gathered feat) + pe
    for (int col = tid; col < COLS; col += 256) {
        int p = col / KNB, j = col - p * KNB;
        int nb = kb[p * KNB + j];
        _Float16* a = A0 + (size_t)col * C;
        const _Float16* pe = PE + (size_t)col * C;
        const float* gf = fbb + (size_t)nb * C;
        for (int c = 0; c < C; ++c)
            a[c] = (_Float16)((kfb[p * C + c] - gf[c]) + (float)pe[c]);
    }
    __syncthreads();
    mlp_layer<C / 32, 4>(AW1, as1, ah1, A0, H1, 64, true, lane, wave);   // C->64 bn+lrelu
    __syncthreads();
    mlp_layer<2, OCP / 16>(AW2, as2, ah2, H1, ATT, OCP, false, lane, wave); // 64->3+C bias
    __syncthreads();

    // softmax over K + weighted sums
    for (int t = tid; t < OC * TP; t += 256) {
        int oc = t % OC, p = t / OC;
        const int* kn = kb + p * KNB;
        float w[KNB];
        float mx = -3.4e38f;
#pragma unroll
        for (int j = 0; j < KNB; ++j) {
            w[j] = (float)ATT[(size_t)(p * KNB + j) * OCP + oc];
            mx = fmaxf(mx, w[j]);
        }
        float se = 0.f;
#pragma unroll
        for (int j = 0; j < KNB; ++j) { w[j] = __expf(w[j] - mx); se += w[j]; }
        float inv = 1.f / se;
        if (oc < 3) {
            float s = 0.f;
#pragma unroll
            for (int j = 0; j < KNB; ++j) s += w[j] * vb[(size_t)kn[j] * 3 + oc];
            vout[((size_t)b * M + p0 + p) * 3 + oc] = s * inv;
        } else {
            int c = oc - 3;
            float s = 0.f;
#pragma unroll
            for (int j = 0; j < KNB; ++j)
                s += w[j] * (fbb[(size_t)kn[j] * C + c] + (float)PE[(size_t)(p * KNB + j) * C + c]);
            fout[((size_t)b * M + p0 + p) * C + c] = s * inv;
        }
    }
}

// ---------------------------------------------------------------------------
// KNN: one wave per query. Per-lane sorted top-20 lists in LDS, merged with
// 20 rounds of wave-wide argmin via __shfl_xor (wave32).
// query:[B,NQ,C] cand:[B,NC,C] out:[B,NQ,20]
// ---------------------------------------------------------------------------
template<int C>
__global__ __launch_bounds__(256) void knn_kernel(
    const float* __restrict__ query, const float* __restrict__ cand,
    int* __restrict__ out, int NQ, int NC)
{
    __shared__ float qbuf[8][(C < 4) ? 4 : C];
    __shared__ float ldv[8][32 * 20];
    __shared__ int   liv[8][32 * 20];
    int wave = threadIdx.x >> 5, lane = threadIdx.x & 31;
    int gq = blockIdx.x * 8 + wave;
    int b = gq / NQ, q = gq - b * NQ;
    const float* qp = query + ((size_t)b * NQ + q) * C;
    for (int c = lane; c < C; c += 32) qbuf[wave][c] = qp[c];
    __builtin_amdgcn_wave_barrier();
    float* myd = ldv[wave] + lane * 20;
    int*   myi = liv[wave] + lane * 20;
#pragma unroll
    for (int t = 0; t < 20; ++t) { myd[t] = 3.4e38f; myi[t] = 0x7fffffff; }
    const float* cb = cand + (size_t)b * NC * C;
    for (int n = lane; n < NC; n += 32) {
        const float* cp = cb + (size_t)n * C;
        float d = 0.f;
#pragma unroll
        for (int c = 0; c < C; ++c) { float t = qbuf[wave][c] - cp[c]; d += t * t; }
        if (d < myd[19]) {
            int pn = 19;
            while (pn > 0 && myd[pn - 1] > d) { myd[pn] = myd[pn - 1]; myi[pn] = myi[pn - 1]; --pn; }
            myd[pn] = d; myi[pn] = n;
        }
    }
    __builtin_amdgcn_wave_barrier();
    int head = 0;
    int* op = out + ((size_t)b * NQ + q) * 20;
    for (int t = 0; t < 20; ++t) {
        float hd = (head < 20) ? myd[head] : 3.4e38f;
        int   hi = (head < 20) ? myi[head] : 0x7fffffff;
        float rd = hd; int ridx = hi;
        for (int off = 16; off; off >>= 1) {
            float od = __shfl_xor(rd, off);
            int   oi = __shfl_xor(ridx, off);
            if (od < rd || (od == rd && oi < ridx)) { rd = od; ridx = oi; }
        }
        if (head < 20 && hi == ridx) head++;
        if (lane == t) op[t] = ridx;
    }
}

// ---------------------------------------------------------------------------
// Furthest point sampling (matches reference emit order: idx[i] = prev far).
// ---------------------------------------------------------------------------
__global__ __launch_bounds__(256) void fps_kernel(
    const float* __restrict__ pts, int* __restrict__ fidx, int N, int M)
{
    extern __shared__ char smem[];
    float* dist = (float*)smem;
    float* rv = dist + N;
    int*   ri = (int*)(rv + 256);
    int tid = threadIdx.x, b = blockIdx.x;
    const float* pb = pts + (size_t)b * N * 3;
    for (int n = tid; n < N; n += 256) dist[n] = 1e10f;
    __syncthreads();
    int far = 0;
    for (int i = 0; i < M; ++i) {
        if (tid == 0) fidx[(size_t)b * M + i] = far;
        float fx = pb[(size_t)far * 3], fy = pb[(size_t)far * 3 + 1], fz = pb[(size_t)far * 3 + 2];
        float bv = -1.f; int bi = 0x7fffffff;
        for (int n = tid; n < N; n += 256) {
            float dx = pb[(size_t)n * 3] - fx, dy = pb[(size_t)n * 3 + 1] - fy,
                  dz = pb[(size_t)n * 3 + 2] - fz;
            float d = fminf(dist[n], dx * dx + dy * dy + dz * dz);
            dist[n] = d;
            if (d > bv) { bv = d; bi = n; }
        }
        rv[tid] = bv; ri[tid] = bi;
        __syncthreads();
        for (int s = 128; s > 0; s >>= 1) {
            if (tid < s) {
                float ov = rv[tid + s]; int oi = ri[tid + s];
                if (ov > rv[tid] || (ov == rv[tid] && oi < ri[tid])) { rv[tid] = ov; ri[tid] = oi; }
            }
            __syncthreads();
        }
        far = ri[0];
        __syncthreads();
    }
}

__global__ __launch_bounds__(256) void gather_kernel(
    const float* __restrict__ src, const int* __restrict__ fidx,
    float* __restrict__ dst, int N, int M, int C)
{
    int b = blockIdx.x;
    const float* sb = src + (size_t)b * N * C;
    const int* ib = fidx + (size_t)b * M;
    float* db = dst + (size_t)b * M * C;
    for (int t = threadIdx.x; t < M * C; t += 256) {
        int i = t / C, c = t - i * C;
        db[t] = sb[(size_t)ib[i] * C + c];
    }
}

__global__ __launch_bounds__(256) void final_kernel(
    const float* __restrict__ feat, float* __restrict__ out, int N, int C)
{
    int b = blockIdx.x;
    const float* fb = feat + (size_t)b * N * C;
    for (int c = threadIdx.x; c < C; c += 256) {
        float mx = -3.4e38f, sm = 0.f;
        for (int n = 0; n < N; ++n) {
            float v = fb[(size_t)n * C + c];
            mx = fmaxf(mx, v);
            sm += v;
        }
        out[(size_t)b * 2 * C + c] = mx;
        out[(size_t)b * 2 * C + C + c] = sm / (float)N;
    }
}

// ---------------------------------------------------------------------------
// Parameter prep: pack weights to padded f16; fold BN/bias into scale/shift.
// ---------------------------------------------------------------------------
__global__ __launch_bounds__(256) void pack_w_kernel(
    const float* __restrict__ w, _Float16* __restrict__ dst, int Co, int Ci, int CoP, int CiP)
{
    int t = blockIdx.x * 256 + threadIdx.x;
    if (t < CoP * CiP) {
        int r = t / CiP, c = t - r * CiP;
        dst[t] = (r < Co && c < Ci) ? (_Float16)w[r * Ci + c] : (_Float16)0.f;
    }
}

__global__ __launch_bounds__(256) void fuse_bn_kernel(
    const float* __restrict__ g, const float* __restrict__ bt, const float* __restrict__ m,
    const float* __restrict__ v, const float* __restrict__ b,
    float* __restrict__ scale, float* __restrict__ shift, int C, int CP)
{
    int t = blockIdx.x * 256 + threadIdx.x;
    if (t < CP) {
        if (t < C) {
            float s = g[t] / sqrtf(v[t] + 1e-5f);
            scale[t] = s;
            shift[t] = (b[t] - m[t]) * s + bt[t];
        } else { scale[t] = 0.f; shift[t] = 0.f; }
    }
}

__global__ __launch_bounds__(256) void fuse_bias_kernel(
    const float* __restrict__ b, float* __restrict__ scale, float* __restrict__ shift,
    int C, int CP)
{
    int t = blockIdx.x * 256 + threadIdx.x;
    if (t < CP) { scale[t] = (t < C) ? 1.f : 0.f; shift[t] = (t < C) ? b[t] : 0.f; }
}

// ---------------------------------------------------------------------------
extern "C" void kernel_launch(void* const* d_in, const int* in_sizes, int n_in,
                              void* d_out, int out_size, void* d_ws, size_t ws_size,
                              hipStream_t stream)
{
    (void)in_sizes; (void)n_in; (void)out_size; (void)ws_size;
    const int B = 16, N0 = 4096, M1 = 1024, M2 = 512;
    const float* pos = (const float*)d_in[0];

    int ip = 1;
    auto P = [&]() { return (const float*)d_in[ip++]; };
    // g1
    const float *g1c1w=P(),*g1c1b=P(),*g1b1g=P(),*g1b1b=P(),*g1b1m=P(),*g1b1v=P();
    const float *g1c2w=P(),*g1c2b=P(),*g1b2g=P(),*g1b2b=P(),*g1b2m=P(),*g1b2v=P();
    const float *g1c3w=P(),*g1c3b=P();
    // p1 pos / attn
    const float *p1pc1w=P(),*p1pc1b=P(),*p1pbg=P(),*p1pbb=P(),*p1pbm=P(),*p1pbv=P();
    const float *p1pc2w=P(),*p1pc2b=P();
    const float *p1ac1w=P(),*p1ac1b=P(),*p1abg=P(),*p1abb=P(),*p1abm=P(),*p1abv=P();
    const float *p1ac2w=P(),*p1ac2b=P();
    // g2
    const float *g2c1w=P(),*g2c1b=P(),*g2b1g=P(),*g2b1b=P(),*g2b1m=P(),*g2b1v=P();
    const float *g2c2w=P(),*g2c2b=P(),*g2b2g=P(),*g2b2b=P(),*g2b2m=P(),*g2b2v=P();
    const float *g2c3w=P(),*g2c3b=P();
    // p2 pos / attn
    const float *p2pc1w=P(),*p2pc1b=P(),*p2pbg=P(),*p2pbb=P(),*p2pbm=P(),*p2pbv=P();
    const float *p2pc2w=P(),*p2pc2b=P();
    const float *p2ac1w=P(),*p2ac1b=P(),*p2abg=P(),*p2abb=P(),*p2abm=P(),*p2abv=P();
    const float *p2ac2w=P(),*p2ac2b=P();
    // g3
    const float *g3c1w=P(),*g3c1b=P(),*g3b1g=P(),*g3b1b=P(),*g3b1m=P(),*g3b1v=P();
    const float *g3c2w=P(),*g3c2b=P(),*g3b2g=P(),*g3b2b=P(),*g3b2m=P(),*g3b2v=P();
    const float *g3c3w=P(),*g3c3b=P();

    char* wsp = (char*)d_ws;
    size_t off = 0;
    auto alloc = [&](size_t bytes) -> void* {
        size_t o = (off + 255) & ~(size_t)255;
        off = o + bytes;
        return (void*)(wsp + o);
    };
    auto hal = [&](size_t n) { return (_Float16*)alloc(n * 2); };
    auto fal = [&](size_t n) { return (float*)alloc(n * 4); };
    auto ial = [&](size_t n) { return (int*)alloc(n * 4); };

    // packed f16 weights
    _Float16 *g1W1=hal(32*32), *g1W2=hal(32*32), *g1W3=hal(64*32);
    _Float16 *p1PW1=hal(64*32), *p1PW2=hal(64*64), *p1AW1=hal(64*64), *p1AW2=hal(80*64);
    _Float16 *g2W1=hal(64*128), *g2W2=hal(64*64), *g2W3=hal(128*64);
    _Float16 *p2PW1=hal(64*32), *p2PW2=hal(128*64), *p2AW1=hal(64*128), *p2AW2=hal(144*64);
    _Float16 *g3W1=hal(256*256), *g3W2=hal(256*256), *g3W3=hal(512*256);
    // fused scale/shift
    float *g1s1=fal(32),*g1h1=fal(32),*g1s2=fal(32),*g1h2=fal(32),*g1s3=fal(64),*g1h3=fal(64);
    float *p1ps1=fal(64),*p1ph1=fal(64),*p1ps2=fal(64),*p1ph2=fal(64);
    float *p1as1=fal(64),*p1ah1=fal(64),*p1as2=fal(80),*p1ah2=fal(80);
    float *g2s1=fal(64),*g2h1=fal(64),*g2s2=fal(64),*g2h2=fal(64),*g2s3=fal(128),*g2h3=fal(128);
    float *p2ps1=fal(64),*p2ph1=fal(64),*p2ps2=fal(128),*p2ph2=fal(128);
    float *p2as1=fal(64),*p2ah1=fal(64),*p2as2=fal(144),*p2ah2=fal(144);
    float *g3s1=fal(256),*g3h1=fal(256),*g3s2=fal(256),*g3h2=fal(256),*g3s3=fal(512),*g3h3=fal(512);
    // activations / indices
    float* feat1  = fal((size_t)B*N0*64);
    int*   idxg1  = ial((size_t)B*N0*20);
    int*   fidx1  = ial((size_t)B*M1);
    float* kp1    = fal((size_t)B*M1*3);
    float* kf1    = fal((size_t)B*M1*64);
    int*   idxp1  = ial((size_t)B*M1*20);
    float* v1     = fal((size_t)B*M1*3);
    float* feat1p = fal((size_t)B*M1*64);
    int*   idxg2  = ial((size_t)B*M1*20);
    float* feat2  = fal((size_t)B*M1*128);
    int*   fidx2  = ial((size_t)B*M2);
    float* kp2    = fal((size_t)B*M2*3);
    float* kf2    = fal((size_t)B*M2*128);
    int*   idxp2  = ial((size_t)B*M2*20);
    float* v2     = fal((size_t)B*M2*3);
    float* feat2p = fal((size_t)B*M2*128);
    int*   idxg3  = ial((size_t)B*M2*20);
    float* feat3  = fal((size_t)B*M2*512);

    auto packW = [&](const float* w, _Float16* dst, int Co, int Ci, int CoP, int CiP) {
        int tot = CoP * CiP;
        pack_w_kernel<<<(tot + 255) / 256, 256, 0, stream>>>(w, dst, Co, Ci, CoP, CiP);
    };
    auto fBN = [&](const float* g, const float* bt, const float* m, const float* v,
                   const float* b, float* sc, float* sh, int Cc, int CP) {
        fuse_bn_kernel<<<(CP + 255) / 256, 256, 0, stream>>>(g, bt, m, v, b, sc, sh, Cc, CP);
    };
    auto fB = [&](const float* b, float* sc, float* sh, int Cc, int CP) {
        fuse_bias_kernel<<<(CP + 255) / 256, 256, 0, stream>>>(b, sc, sh, Cc, CP);
    };

    // ---- parameter prep ----
    packW(g1c1w, g1W1, 32, 6, 32, 32);   packW(g1c2w, g1W2, 32, 32, 32, 32);
    packW(g1c3w, g1W3, 64, 32, 64, 32);
    packW(p1pc1w, p1PW1, 64, 3, 64, 32); packW(p1pc2w, p1PW2, 64, 64, 64, 64);
    packW(p1ac1w, p1AW1, 64, 64, 64, 64); packW(p1ac2w, p1AW2, 67, 64, 80, 64);
    packW(g2c1w, g2W1, 64, 128, 64, 128); packW(g2c2w, g2W2, 64, 64, 64, 64);
    packW(g2c3w, g2W3, 128, 64, 128, 64);
    packW(p2pc1w, p2PW1, 64, 3, 64, 32); packW(p2pc2w, p2PW2, 128, 64, 128, 64);
    packW(p2ac1w, p2AW1, 64, 128, 64, 128); packW(p2ac2w, p2AW2, 131, 64, 144, 64);
    packW(g3c1w, g3W1, 256, 256, 256, 256); packW(g3c2w, g3W2, 256, 256, 256, 256);
    packW(g3c3w, g3W3, 512, 256, 512, 256);

    fBN(g1b1g, g1b1b, g1b1m, g1b1v, g1c1b, g1s1, g1h1, 32, 32);
    fBN(g1b2g, g1b2b, g1b2m, g1b2v, g1c2b, g1s2, g1h2, 32, 32);
    fB(g1c3b, g1s3, g1h3, 64, 64);
    fBN(p1pbg, p1pbb, p1pbm, p1pbv, p1pc1b, p1ps1, p1ph1, 64, 64);
    fB(p1pc2b, p1ps2, p1ph2, 64, 64);
    fBN(p1abg, p1abb, p1abm, p1abv, p1ac1b, p1as1, p1ah1, 64, 64);
    fB(p1ac2b, p1as2, p1ah2, 67, 80);
    fBN(g2b1g, g2b1b, g2b1m, g2b1v, g2c1b, g2s1, g2h1, 64, 64);
    fBN(g2b2g, g2b2b, g2b2m, g2b2v, g2c2b, g2s2, g2h2, 64, 64);
    fB(g2c3b, g2s3, g2h3, 128, 128);
    fBN(p2pbg, p2pbb, p2pbm, p2pbv, p2pc1b, p2ps1, p2ph1, 64, 64);
    fB(p2pc2b, p2ps2, p2ph2, 128, 128);
    fBN(p2abg, p2abb, p2abm, p2abv, p2ac1b, p2as1, p2ah1, 64, 64);
    fB(p2ac2b, p2as2, p2ah2, 131, 144);
    fBN(g3b1g, g3b1b, g3b1m, g3b1v, g3c1b, g3s1, g3h1, 256, 256);
    fBN(g3b2g, g3b2b, g3b2m, g3b2v, g3c2b, g3s2, g3h2, 256, 256);
    fB(g3c3b, g3s3, g3h3, 512, 512);

    // LDS sizes
    size_t sm1 = (size_t)(32 + 32) * 160 * 2 + 8 * 16 * 160 * 4;     // 102400
    size_t sm2 = (size_t)(128 + 64) * 160 * 2 + 8 * 16 * 160 * 4;    // 143360
    size_t sm3 = (size_t)(256 + 256) * 160 * 2 + 8 * 16 * 160 * 4;   // 245760
    size_t sa1 = (size_t)(32 + 64 + 64 + 64 + 80) * 160 * 2;         // 97280
    size_t sa2 = (size_t)(32 + 64 + 128 + 128 + 144) * 160 * 2;      // 158720
    size_t sf1 = (size_t)N0 * 4 + 256 * 8 + 16;
    size_t sf2 = (size_t)M1 * 4 + 256 * 8 + 16;

    // ---- stage 1: EdgeConv g1 (xyz features) ----
    knn_kernel<3><<<B * N0 / 8, 256, 0, stream>>>(pos, pos, idxg1, N0, N0);
    edgeconv_kernel<3, 32, 64, 4096><<<B * (N0 / 8), 256, sm1, stream>>>(
        pos, idxg1, g1W1, g1s1, g1h1, g1W2, g1s2, g1h2, g1W3, g1s3, g1h3, feat1);

    // ---- AGP 1 ----
    fps_kernel<<<B, 256, sf1, stream>>>(pos, fidx1, N0, M1);
    gather_kernel<<<B, 256, 0, stream>>>(pos, fidx1, kp1, N0, M1, 3);
    gather_kernel<<<B, 256, 0, stream>>>(feat1, fidx1, kf1, N0, M1, 64);
    knn_kernel<3><<<B * M1 / 8, 256, 0, stream>>>(kp1, pos, idxp1, M1, N0);
    agp_kernel<64><<<B * (M1 / 8), 256, sa1, stream>>>(
        pos, feat1, kp1, kf1, idxp1,
        p1PW1, p1ps1, p1ph1, p1PW2, p1ps2, p1ph2,
        p1AW1, p1as1, p1ah1, p1AW2, p1as2, p1ah2, v1, feat1p, N0, M1);

    // ---- stage 2: EdgeConv g2 (64-d feature KNN) ----
    knn_kernel<64><<<B * M1 / 8, 256, 0, stream>>>(feat1p, feat1p, idxg2, M1, M1);
    edgeconv_kernel<64, 64, 128, 1024><<<B * (M1 / 8), 256, sm2, stream>>>(
        feat1p, idxg2, g2W1, g2s1, g2h1, g2W2, g2s2, g2h2, g2W3, g2s3, g2h3, feat2);

    // ---- AGP 2 ----
    fps_kernel<<<B, 256, sf2, stream>>>(v1, fidx2, M1, M2);
    gather_kernel<<<B, 256, 0, stream>>>(v1, fidx2, kp2, M1, M2, 3);
    gather_kernel<<<B, 256, 0, stream>>>(feat2, fidx2, kf2, M1, M2, 128);
    knn_kernel<3><<<B * M2 / 8, 256, 0, stream>>>(kp2, v1, idxp2, M2, M1);
    agp_kernel<128><<<B * (M2 / 8), 256, sa2, stream>>>(
        v1, feat2, kp2, kf2, idxp2,
        p2PW1, p2ps1, p2ph1, p2PW2, p2ps2, p2ph2,
        p2AW1, p2as1, p2ah1, p2AW2, p2as2, p2ah2, v2, feat2p, M1, M2);

    // ---- stage 3: EdgeConv g3 (128-d feature KNN) ----
    knn_kernel<128><<<B * M2 / 8, 256, 0, stream>>>(feat2p, feat2p, idxg3, M2, M2);
    edgeconv_kernel<128, 256, 512, 512><<<B * (M2 / 8), 256, sm3, stream>>>(
        feat2p, idxg3, g3W1, g3s1, g3h1, g3W2, g3s2, g3h2, g3W3, g3s3, g3h3, feat3);

    // ---- global max + mean pooling ----
    final_kernel<<<B, 256, 0, stream>>>(feat3, (float*)d_out, M2, 512);
}
